// MyModel_12738873000491
// MI455X (gfx1250) — matrix operand order repaired and checked by reading, not compiled, not run on hardware
//
#include <hip/hip_runtime.h>
#include <hip/hip_bf16.h>

typedef __attribute__((ext_vector_type(16))) _Float16 v16h;
typedef __attribute__((ext_vector_type(8)))  float    v8f;
typedef __attribute__((ext_vector_type(4)))  float    f32x4;

#define B_ROWS 8192
#define H      256

// ---------------------------------------------------------------------------
// Kernel 0: swizzle the two 256x256 f32 hidden-layer weights into f16
// WMMA B-fragment order:  BW[ ((nt*8 + kk)*32 + lane)*16 + e ]
//   n = nt*16 + (lane&15),  k = kk*32 + (lane<16 ? 0 : 16) + e
// so the GEMM loop loads one contiguous 32-byte v16h per lane (coalesced).
// ---------------------------------------------------------------------------
__global__ void bprep_kernel(const float* __restrict__ W3b,
                             const float* __restrict__ W4b,
                             _Float16* __restrict__ BW) {
    int idx  = blockIdx.x * blockDim.x + threadIdx.x;   // 0 .. 131071
    int net  = idx >> 16;
    int r    = idx & 65535;
    int e    = r & 15;
    int lane = (r >> 4) & 31;
    int kk   = (r >> 9) & 7;
    int nt   = r >> 12;
    int n    = nt * 16 + (lane & 15);
    int k    = kk * 32 + ((lane < 16) ? 0 : 16) + e;
    const float* W = net ? W4b : W3b;
    BW[idx] = (_Float16)W[k * H + n];
}

// ---------------------------------------------------------------------------
// Kernel 1: per-column j: bilinear interp com_speed_min[j], contiguous
// copy of last_ac[j] (x is row-major stride 7).
// ---------------------------------------------------------------------------
__global__ void colprep_kernel(const float* __restrict__ x,
                               const float* __restrict__ lo_t,
                               const float* __restrict__ hi_t,
                               const float* __restrict__ csm_t,
                               float* __restrict__ csm,
                               float* __restrict__ lastac) {
    int j = blockIdx.x * blockDim.x + threadIdx.x;
    if (j >= B_ROWS) return;
    float lo_p = x[j * 7 + 1];
    float hi_p = x[j * 7 + 2];
    int ci = 0, cj = 0;
    #pragma unroll
    for (int k = 0; k < 10; ++k) {            // searchsorted(side='left')
        ci += (lo_t[k] < lo_p);
        cj += (hi_t[k] < hi_p);
    }
    int i  = min(max(ci - 1, 0), 8);
    int jj = min(max(cj - 1, 0), 8);
    float Q11 = csm_t[i * 10 + jj];
    float Q12 = csm_t[i * 10 + jj + 1];
    float Q21 = csm_t[(i + 1) * 10 + jj];
    float Q22 = csm_t[(i + 1) * 10 + jj + 1];
    float xr = (lo_p - lo_t[i])  / (lo_t[i + 1]  - lo_t[i]);
    float yr = (hi_p - hi_t[jj]) / (hi_t[jj + 1] - hi_t[jj]);
    float R1 = xr * (Q21 - Q11) + Q11;
    float R2 = xr * (Q22 - Q12) + Q12;
    csm[j]    = yr * (R2 - R1) + R1;
    lastac[j] = x[j * 7 + 0];
}

// ---------------------------------------------------------------------------
// Kernel 2: fused dual-MLP.  One block = 16 batch rows, 256 threads = 8 waves.
// Layer 1 (K=6 / K=2) on VALU -> tanh -> f16 into LDS (A operand).
// Layer 2 (256x256) via v_wmma_f32_16x16x32_f16: wave w owns N-tiles 2w,2w+1,
// 8 WMMAs each over K=256.  Layer 3 (H->1) folded into the epilogue:
// tanh(D + b) * Wc accumulated per-row with LDS float atomics.
// Output: s[i] = part1[i] + part2[i] + b3c + b4c.
// ---------------------------------------------------------------------------
__global__ void __launch_bounds__(256)
mlp_kernel(const float* __restrict__ x,
           const float* __restrict__ W3a, const float* __restrict__ b3a,
           const float* __restrict__ b3b, const float* __restrict__ W3c,
           const float* __restrict__ b3c,
           const float* __restrict__ W4a, const float* __restrict__ b4a,
           const float* __restrict__ b4b, const float* __restrict__ W4c,
           const float* __restrict__ b4c,
           const _Float16* __restrict__ BW,
           float* __restrict__ s_arr) {
    __shared__ float    feat[16][8];      // 16 rows x 7 raw features (padded)
    __shared__ _Float16 hA[16 * H];       // layer-1 activations, f16, 8 KB
    __shared__ float    partial[16];      // per-row accumulator for layer 3

    const int tid  = threadIdx.x;
    const int r0   = blockIdx.x * 16;
    const int lane = tid & 31;
    const int wave = tid >> 5;
    const int m    = lane & 15;

    if (tid < 112) {                      // load 16 rows x 7 features
        int row = tid / 7, k = tid % 7;
        feat[row][k] = x[(size_t)(r0 + row) * 7 + k];
    }
    if (tid < 16) partial[tid] = 0.0f;
    __syncthreads();

    const int basek_a = (lane < 16) ? 0 : 16;         // A-fragment K base
    const int nt0 = wave * 2, nt1 = wave * 2 + 1;     // this wave's N tiles
    const int rowbase = (lane < 16) ? 0 : 8;          // D-fragment M base

    #pragma unroll 1
    for (int net = 0; net < 2; ++net) {
        // ------- layer 1: thread t owns column c = t for all 16 rows -------
        const int c = tid;
        if (net == 0) {
            float w0 = W3a[0 * H + c], w1 = W3a[1 * H + c], w2 = W3a[2 * H + c];
            float w3 = W3a[3 * H + c], w4 = W3a[4 * H + c], w5 = W3a[5 * H + c];
            float bias = b3a[c];
            #pragma unroll
            for (int i = 0; i < 16; ++i) {
                const float* f = feat[i];
                // use_x1 = [t2, t5, hi_p, t1, lo_p, aim_hp]
                float acc = bias + w0 * f[4] + w1 * f[6] + w2 * f[2]
                                 + w3 * f[5] + w4 * f[1] + w5 * f[3];
                hA[i * H + c] = (_Float16)tanhf(acc);
            }
        } else {
            float w0 = W4a[0 * H + c], w1 = W4a[1 * H + c];
            float bias = b4a[c];
            #pragma unroll
            for (int i = 0; i < 16; ++i) {
                const float* f = feat[i];
                float dif_t   = f[4] - f[5];          // t2 - t1
                float diff_hp = f[3] - f[2];          // aim_hp - hi_p
                float acc = bias + w0 * dif_t + w1 * diff_hp;
                hA[i * H + c] = (_Float16)tanhf(acc);
            }
        }
        __syncthreads();

        // ------- layer 2: WMMA over K = 256 (8 chunks of 32) -------
        const _Float16* BWn = BW + (size_t)net * 65536;
        v8f c0 = {}, c1 = {};
        #pragma unroll
        for (int kk = 0; kk < 8; ++kk) {
            v16h a;
            const _Float16* ap = &hA[m * H + kk * 32 + basek_a];
            #pragma unroll
            for (int e = 0; e < 16; ++e)               // K = basek + e
                a[e] = ap[e];
            v16h b0 = *(const v16h*)&BWn[((size_t)(nt0 * 8 + kk) * 32 + lane) * 16];
            v16h b1 = *(const v16h*)&BWn[((size_t)(nt1 * 8 + kk) * 32 + lane) * 16];
            c0 = __builtin_amdgcn_wmma_f32_16x16x32_f16(false, a, false, b0,
                                                        (short)0, c0, false, false);
            c1 = __builtin_amdgcn_wmma_f32_16x16x32_f16(false, a, false, b1,
                                                        (short)0, c1, false, false);
        }

        // ------- layer 3 epilogue: tanh(D + b) * Wc, reduce per row -------
        const float* bb = net ? b4b : b3b;
        const float* wc = net ? W4c : W3c;
        const int col0 = nt0 * 16 + m, col1 = nt1 * 16 + m;
        float bb0 = bb[col0], bb1 = bb[col1];
        float wc0 = wc[col0], wc1 = wc[col1];
        #pragma unroll
        for (int r = 0; r < 8; ++r) {
            float t0 = tanhf(c0[r] + bb0);
            float t1 = tanhf(c1[r] + bb1);
            atomicAdd(&partial[rowbase + r], t0 * wc0 + t1 * wc1);
        }
        __syncthreads();   // partial done; hA free for next net
    }

    if (tid < 16)
        s_arr[r0 + tid] = partial[tid] + b3c[0] + b4c[0];
}

// ---------------------------------------------------------------------------
// Kernel 3: the bandwidth kernel.  out[i,j] = min(max(la[j]+s[i], csm[j]),8000)
// 4 rows per block, native f32x4 vectors, non-temporal stores (256 MiB).
// ---------------------------------------------------------------------------
__global__ void __launch_bounds__(256)
bcast_kernel(const float* __restrict__ s,
             const float* __restrict__ csm,
             const float* __restrict__ la,
             float* __restrict__ out) {
    const int i0 = blockIdx.x * 4;
    float sv[4];
    #pragma unroll
    for (int r = 0; r < 4; ++r) sv[r] = s[i0 + r];

    const f32x4* la4 = (const f32x4*)la;
    const f32x4* cs4 = (const f32x4*)csm;
    f32x4*       o4  = (f32x4*)out;

    #pragma unroll
    for (int it = 0; it < 8; ++it) {
        int j4 = it * 256 + threadIdx.x;          // 2048 f32x4 per row
        f32x4 lav = la4[j4];
        f32x4 cs  = cs4[j4];
        #pragma unroll
        for (int r = 0; r < 4; ++r) {
            f32x4 o;
            o.x = fminf(fmaxf(lav.x + sv[r], cs.x), 8000.0f);
            o.y = fminf(fmaxf(lav.y + sv[r], cs.y), 8000.0f);
            o.z = fminf(fmaxf(lav.z + sv[r], cs.z), 8000.0f);
            o.w = fminf(fmaxf(lav.w + sv[r], cs.w), 8000.0f);
            __builtin_nontemporal_store(o, &o4[(size_t)(i0 + r) * 2048 + j4]);
        }
    }
}

// ---------------------------------------------------------------------------
extern "C" void kernel_launch(void* const* d_in, const int* in_sizes, int n_in,
                              void* d_out, int out_size, void* d_ws, size_t ws_size,
                              hipStream_t stream) {
    const float* x    = (const float*)d_in[0];
    const float* W3a  = (const float*)d_in[1];
    const float* b3a  = (const float*)d_in[2];
    const float* W3b  = (const float*)d_in[3];
    const float* b3b  = (const float*)d_in[4];
    const float* W3c  = (const float*)d_in[5];
    const float* b3c  = (const float*)d_in[6];
    const float* W4a  = (const float*)d_in[7];
    const float* b4a  = (const float*)d_in[8];
    const float* W4b  = (const float*)d_in[9];
    const float* b4b  = (const float*)d_in[10];
    const float* W4c  = (const float*)d_in[11];
    const float* b4c  = (const float*)d_in[12];
    const float* lo_t = (const float*)d_in[13];
    const float* hi_t = (const float*)d_in[14];
    const float* csmT = (const float*)d_in[15];

    char* ws = (char*)d_ws;
    _Float16* BW     = (_Float16*)ws;                 // 2 x 128 KB f16 weights
    float*    s_arr  = (float*)(ws + 262144);         // 32 KB
    float*    csm    = (float*)(ws + 294912);         // 32 KB
    float*    lastac = (float*)(ws + 327680);         // 32 KB

    bprep_kernel  <<<512,  256, 0, stream>>>(W3b, W4b, BW);
    colprep_kernel<<<32,   256, 0, stream>>>(x, lo_t, hi_t, csmT, csm, lastac);
    mlp_kernel    <<<512,  256, 0, stream>>>(x, W3a, b3a, b3b, W3c, b3c,
                                             W4a, b4a, b4b, W4c, b4c, BW, s_arr);
    bcast_kernel  <<<2048, 256, 0, stream>>>(s_arr, csm, lastac, (float*)d_out);
}
